// Net_41824391529215
// MI455X (gfx1250) — matrix operand order repaired and checked by reading, not compiled, not run on hardware
//
#include <hip/hip_runtime.h>
#include <hip/hip_bf16.h>

// Problem constants (from reference)
#define B_TOT   256
#define SEQ_L   256
#define DM      256
#define DI      512
#define DS_N    16
#define DR_N    16
#define NLAY    6
#define MLPH    128
#define CB      32              // sequences per chunk (working set fits in 192MB L2)
#define RCH     (CB * SEQ_L)    // 8192 rows per chunk
#define NCHUNK  (B_TOT / CB)

// GEMM tiling
#define BM      64              // block tile M (2 waves x 32)
#define BN      256             // block tile N (4 waves x 64)
#define KSTEP   64              // K slab staged per TDM transfer

typedef __attribute__((ext_vector_type(16))) _Float16 v16h;
typedef __attribute__((ext_vector_type(8)))  _Float16 v8h;
typedef __attribute__((ext_vector_type(8)))  float    v8f;
typedef __attribute__((ext_vector_type(4)))  unsigned int u32x4;
typedef __attribute__((ext_vector_type(8)))  int i32x8;
typedef __attribute__((ext_vector_type(4)))  int i32x4;

// ---------------------------------------------------------------------------
// Tensor Data Mover: 2D tile (rows x kelems of f16) global -> LDS.
// D# per CDNA5 ISA 8.3/8.4: group0 = {count|flags, lds_addr, global_addr, type},
// group1 = {mask/data_size, tensor dims, tile dims, dim0 stride}. Groups 2/3
// zero (2D tensor). Tensor dims = remaining extent from tile start so the
// in-bounds tile never trips OOB; dim0 stride = full row pitch.
// ---------------------------------------------------------------------------
__device__ __forceinline__ void tdm_load_2d(unsigned lds_off, const _Float16* gptr,
                                            unsigned rem_elems, unsigned rem_rows,
                                            unsigned long long row_pitch,
                                            unsigned tile_k, unsigned tile_rows)
{
    unsigned long long ga = (unsigned long long)(uintptr_t)gptr;
    u32x4 g0;
    g0[0] = 1u;                                   // count=1, user descriptor
    g0[1] = lds_off;                              // lds_addr (bytes)
    g0[2] = (unsigned)(ga & 0xffffffffu);         // global_addr[31:0]
    g0[3] = (unsigned)((ga >> 32) & 0x01ffffffu)  // global_addr[56:32]
          | (2u << 30);                           // type = 2 ("image")
    i32x8 g1 = (i32x8)0;
    g1[0] = (int)(1u << 16);                      // data_size = 1 (2 bytes/elem)
    g1[1] = (int)((rem_elems & 0xffffu) << 16);                       // dim0[15:0]
    g1[2] = (int)(((rem_elems >> 16) & 0xffffu) | ((rem_rows & 0xffffu) << 16));
    g1[3] = (int)(((rem_rows >> 16) & 0xffffu) | ((tile_k & 0xffffu) << 16));
    g1[4] = (int)(tile_rows & 0xffffu);           // tile_dim1 rows, tile_dim2=0
    g1[5] = (int)(unsigned)(row_pitch & 0xffffffffu);                 // stride[31:0]
    g1[6] = (int)(unsigned)((row_pitch >> 32) & 0xffffu);             // stride[47:32]
    i32x4 z4 = (i32x4)0;
#if defined(__clang_major__) && (__clang_major__ >= 23)
    i32x8 z8 = (i32x8)0;
    __builtin_amdgcn_tensor_load_to_lds(g0, g1, z4, z4, z8, 0);
#else
    __builtin_amdgcn_tensor_load_to_lds(g0, g1, z4, z4, 0);
#endif
}

__device__ __forceinline__ v16h make_frag(const _Float16* p, int second_off) {
    v8h x0 = *(const v8h*)p;
    v8h x1 = *(const v8h*)(p + second_off);
    return __builtin_shufflevector(x0, x1, 0,1,2,3,4,5,6,7,8,9,10,11,12,13,14,15);
}

// ---------------------------------------------------------------------------
// GEMM: C[M,N] = A[M,K] * W[N,K]^T, f16 inputs, f32 accum.
// Block = 64x256 (8 waves, each a 32x64 register tile = 2x4 WMMA accums).
// A/B slabs double-buffered in LDS via TDM; fragments read with ds_load_b128.
// epilogues: 0 = store f32, 1 = store f16, 2 = f32 += , 3 = softplus(x+bias)
// ---------------------------------------------------------------------------
__global__ __launch_bounds__(256) void wmma_gemm_kernel(
    const _Float16* __restrict__ A, int lda,
    const _Float16* __restrict__ W, int ldw,
    int M, int N, int K,
    float* __restrict__ Cf, _Float16* __restrict__ Ch, int ldc,
    const float* __restrict__ bias, int mode)
{
    __shared__ alignas(16) _Float16 ldsA[2][BM * KSTEP];   // 2 x 8 KB
    __shared__ alignas(16) _Float16 ldsB[2][BN * KSTEP];   // 2 x 32 KB

    int lane = threadIdx.x & 31;
    int wave = threadIdx.x >> 5;
    int nbN = (N + BN - 1) / BN;
    int bm = blockIdx.x / nbN, bn = blockIdx.x % nbN;
    int M0 = bm * BM, N0 = bn * BN;
    int wm = wave >> 2, wn = wave & 3;               // 2 x 4 wave grid
    int rows_n = N - N0; if (rows_n > BN) rows_n = BN;

    v8f acc[2][4];
#pragma unroll
    for (int im = 0; im < 2; ++im)
#pragma unroll
        for (int in = 0; in < 4; ++in) acc[im][in] = {};

    int nsteps = (K + KSTEP - 1) / KSTEP;
    int kc0 = (K < KSTEP) ? K : KSTEP;
    if (wave == 0) {
        tdm_load_2d((unsigned)(uintptr_t)(void*)&ldsA[0][0], A + (size_t)M0 * lda,
                    (unsigned)K, (unsigned)(M - M0), (unsigned long long)lda,
                    (unsigned)kc0, BM);
        tdm_load_2d((unsigned)(uintptr_t)(void*)&ldsB[0][0], W + (size_t)N0 * ldw,
                    (unsigned)K, (unsigned)(N - N0), (unsigned long long)ldw,
                    (unsigned)kc0, (unsigned)rows_n);
    }

    for (int s = 0; s < nsteps; ++s) {
        int k0 = s * KSTEP;
        int k_cur = K - k0; if (k_cur > KSTEP) k_cur = KSTEP;
        int cur = s & 1, nxt = cur ^ 1;

        __builtin_amdgcn_s_wait_tensorcnt(0);   // slab `cur` landed (wave0 tracks)
        __syncthreads();                        // visible to all; prev compute done

        if (wave == 0 && s + 1 < nsteps) {      // overlap: prefetch next slab
            int k1 = k0 + KSTEP;
            int k_nxt = K - k1; if (k_nxt > KSTEP) k_nxt = KSTEP;
            tdm_load_2d((unsigned)(uintptr_t)(void*)&ldsA[nxt][0],
                        A + (size_t)M0 * lda + k1, (unsigned)(K - k1),
                        (unsigned)(M - M0), (unsigned long long)lda,
                        (unsigned)k_nxt, BM);
            tdm_load_2d((unsigned)(uintptr_t)(void*)&ldsB[nxt][0],
                        W + (size_t)N0 * ldw + k1, (unsigned)(K - k1),
                        (unsigned)(N - N0), (unsigned long long)ldw,
                        (unsigned)k_nxt, (unsigned)rows_n);
        }

        const _Float16* Ab = &ldsA[cur][0];
        const _Float16* Bb = &ldsB[cur][0];
        int g = lane >> 4, ln = lane & 15;
        for (int ks = 0; ks < k_cur; ks += 32) {
            v16h af[2], bf[4];
#pragma unroll
            for (int im = 0; im < 2; ++im) {
                int row = wm * 32 + im * 16 + ln;
                af[im] = make_frag(Ab + row * k_cur + ks + 8 * g, 16);
            }
#pragma unroll
            for (int in = 0; in < 4; ++in) {
                int col = wn * 64 + in * 16 + ln;
                bf[in] = make_frag(Bb + col * k_cur + ks + 8 * g, 16);
            }
#pragma unroll
            for (int im = 0; im < 2; ++im)
#pragma unroll
                for (int in = 0; in < 4; ++in)
                    acc[im][in] = __builtin_amdgcn_wmma_f32_16x16x32_f16(
                        false, af[im], false, bf[in], (short)0, acc[im][in],
                        false, false);
        }
    }

    // epilogue (C/D layout: lane = n + 16g, vgpr r -> m = r + 8g)
    int g = lane >> 4, ln = lane & 15;
#pragma unroll
    for (int im = 0; im < 2; ++im) {
#pragma unroll
        for (int in = 0; in < 4; ++in) {
            int n = N0 + wn * 64 + in * 16 + ln;
            if (n >= N) continue;
#pragma unroll
            for (int r = 0; r < 8; ++r) {
                int m = M0 + wm * 32 + im * 16 + r + 8 * g;
                size_t idx = (size_t)m * ldc + n;
                float v = acc[im][in][r];
                if (mode == 0) {
                    Cf[idx] = v;
                } else if (mode == 1) {
                    Ch[idx] = (_Float16)v;
                } else if (mode == 2) {
                    Cf[idx] += v;
                } else {
                    float x = v + bias[n];
                    Cf[idx] = (x > 20.f) ? x : log1pf(__expf(x));
                }
            }
        }
    }
}

// ---------------------------------------------------------------------------
// Embedding gather + conv3 (pad 1/1) + BN affine + ReLU -> hidden & residual
// ---------------------------------------------------------------------------
__global__ __launch_bounds__(256) void embed_conv_kernel(
    const int* __restrict__ tok, const float* __restrict__ emb,
    const float* __restrict__ conv_w, const float* __restrict__ gamma,
    const float* __restrict__ beta, float* __restrict__ hidden,
    float* __restrict__ residual, int row_base)
{
    int tid = blockIdx.x * 256 + threadIdx.x;
    if (tid >= RCH * DM) return;
    int o = tid & (DM - 1);
    int r = tid >> 8;
    int l = r & (SEQ_L - 1);
    float acc = 0.f;
#pragma unroll
    for (int k = 0; k < 3; ++k) {
        int ll = l + k - 1;
        if (ll < 0 || ll >= SEQ_L) continue;
        int tkn = tok[row_base + r + k - 1];
        const float* e = emb + (size_t)tkn * 64;
        const float* w = conv_w + (size_t)o * 192 + k;   // conv_w[o][i][k]
#pragma unroll 8
        for (int i = 0; i < 64; ++i) acc += e[i] * w[i * 3];
    }
    float y = acc * (gamma[o] * rsqrtf(1.001f)) + beta[o];
    y = fmaxf(y, 0.f);
    hidden[(size_t)r * DM + o]   = y;
    residual[(size_t)r * DM + o] = y;
}

__global__ __launch_bounds__(256) void add_kernel(float* __restrict__ dst,
                                                  const float* __restrict__ src, int n)
{
    int i = blockIdx.x * 256 + threadIdx.x;
    if (i < n) dst[i] += src[i];
}

// ---------------------------------------------------------------------------
// RMSNorm row of 256 -> f16 (one wave per row, wave32 shfl reduction)
// ---------------------------------------------------------------------------
__global__ __launch_bounds__(256) void rmsnorm_f16_kernel(
    const float* __restrict__ x, const float* __restrict__ w,
    _Float16* __restrict__ out, int rows)
{
    int wave = threadIdx.x >> 5, lane = threadIdx.x & 31;
    int row = blockIdx.x * 8 + wave;
    if (row >= rows) return;
    const float* xr = x + (size_t)row * DM;
    float v[8], ss = 0.f;
#pragma unroll
    for (int j = 0; j < 8; ++j) { v[j] = xr[lane + 32 * j]; ss += v[j] * v[j]; }
#pragma unroll
    for (int o = 16; o; o >>= 1) ss += __shfl_xor(ss, o, 32);
    float sc = rsqrtf(ss * (1.f / DM) + 1e-4f);
#pragma unroll
    for (int j = 0; j < 8; ++j) {
        int c = lane + 32 * j;
        out[(size_t)row * DM + c] = (_Float16)(v[j] * sc * w[c]);
    }
}

// ---------------------------------------------------------------------------
// Depthwise causal conv4 + bias + SiLU on xi = xz[:, :DI]; stores f32 + f16
// ---------------------------------------------------------------------------
__global__ __launch_bounds__(256) void conv1d_silu_kernel(
    const _Float16* __restrict__ xz16, const float* __restrict__ cw,
    const float* __restrict__ cb, float* __restrict__ xc,
    _Float16* __restrict__ xc16, int layer)
{
    int tid = blockIdx.x * 256 + threadIdx.x;
    if (tid >= RCH * DI) return;
    int d = tid & (DI - 1);
    int r = tid >> 9;
    int l = r & (SEQ_L - 1);
    const float* wk = cw + ((size_t)layer * DI + d) * 4;
    float acc = cb[layer * DI + d];
#pragma unroll
    for (int k = 0; k < 4; ++k) {
        int ll = l - 3 + k;
        if (ll >= 0) acc += wk[k] * (float)xz16[(size_t)(r - 3 + k) * (2 * DI) + d];
    }
    float v = acc / (1.f + __expf(-acc));
    xc[(size_t)r * DI + d]   = v;
    xc16[(size_t)r * DI + d] = (_Float16)v;
}

__global__ __launch_bounds__(256) void dt_prep_kernel(
    const float* __restrict__ xdbl, _Float16* __restrict__ dtin, int rows)
{
    int tid = blockIdx.x * 256 + threadIdx.x;
    if (tid >= rows * 32) return;
    int j = tid & 31, r = tid >> 5;
    dtin[tid] = (j < DR_N) ? (_Float16)xdbl[(size_t)r * 48 + j] : (_Float16)0.f;
}

// ---------------------------------------------------------------------------
// Selective scan: one thread per (sequence, channel); 256 dependent steps,
// 16 states in registers. Fused SiLU(z) gate + D skip, f16 output.
// ---------------------------------------------------------------------------
__global__ __launch_bounds__(256) void scan_kernel(
    const float* __restrict__ dt, const float* __restrict__ xc,
    const float* __restrict__ xdbl, const _Float16* __restrict__ xz16,
    const float* __restrict__ A_log, const float* __restrict__ Dp,
    _Float16* __restrict__ y16, int layer)
{
    int tid = blockIdx.x * 256 + threadIdx.x;
    if (tid >= CB * DI) return;
    int d = tid & (DI - 1);
    int s = tid >> 9;
    float Ac[DS_N], h[DS_N];
#pragma unroll
    for (int n = 0; n < DS_N; ++n) {
        Ac[n] = -__expf(A_log[((size_t)layer * DI + d) * DS_N + n]);
        h[n] = 0.f;
    }
    float Dv = Dp[layer * DI + d];
    for (int l = 0; l < SEQ_L; ++l) {
        size_t r = (size_t)s * SEQ_L + l;
        float dtv = dt[r * DI + d];
        float u   = xc[r * DI + d];
        const float* Bm = xdbl + r * 48 + DR_N;
        const float* Cm = xdbl + r * 48 + DR_N + DS_N;
        float du = dtv * u, y = 0.f;
#pragma unroll
        for (int n = 0; n < DS_N; ++n) {
            h[n] = __expf(dtv * Ac[n]) * h[n] + du * Bm[n];
            y += h[n] * Cm[n];
        }
        float z = (float)xz16[r * (2 * DI) + DI + d];
        float out = (y + u * Dv) * (z / (1.f + __expf(-z)));
        y16[r * DI + d] = (_Float16)out;
    }
}

__global__ __launch_bounds__(256) void swiglu_kernel(
    const float* __restrict__ yg, _Float16* __restrict__ h16, int rows)
{
    int tid = blockIdx.x * 256 + threadIdx.x;
    if (tid >= rows * MLPH) return;
    int j = tid & (MLPH - 1), r = tid >> 7;
    float a = yg[(size_t)r * 256 + j];
    float g = yg[(size_t)r * 256 + MLPH + j];
    h16[tid] = (_Float16)(a * (g / (1.f + __expf(-g))));
}

// ---------------------------------------------------------------------------
// Final: LayerNorm(hidden+residual), masked mean pool, bind head, sigmoid.
// One block per sequence; deterministic LDS combine.
// ---------------------------------------------------------------------------
__global__ __launch_bounds__(256) void final_pool_kernel(
    const float* __restrict__ hidden, const float* __restrict__ residual,
    const int* __restrict__ mask, const float* __restrict__ nw,
    const float* __restrict__ nb, const float* __restrict__ bw,
    const float* __restrict__ bb, float* __restrict__ out, int chunk)
{
    __shared__ float pool_w[8][DM];
    __shared__ float cnt_w[8];
    __shared__ float red[DM];
    __shared__ float bind_s[3];
    int s = blockIdx.x;
    int wave = threadIdx.x >> 5, lane = threadIdx.x & 31;
    int b = chunk * CB + s;

    float pacc[8];
#pragma unroll
    for (int j = 0; j < 8; ++j) pacc[j] = 0.f;
    float mcnt = 0.f;

    for (int l = wave; l < SEQ_L; l += 8) {
        size_t r = (size_t)s * SEQ_L + l;
        float v[8], sum = 0.f, sq = 0.f;
#pragma unroll
        for (int j = 0; j < 8; ++j) {
            int c = lane + 32 * j;
            v[j] = hidden[r * DM + c] + residual[r * DM + c];
            sum += v[j]; sq += v[j] * v[j];
        }
#pragma unroll
        for (int o = 16; o; o >>= 1) { sum += __shfl_xor(sum, o, 32); sq += __shfl_xor(sq, o, 32); }
        float mu  = sum * (1.f / DM);
        float inv = rsqrtf(sq * (1.f / DM) - mu * mu + 1e-4f);
        float m = (float)mask[(size_t)b * SEQ_L + l];
        if (lane == 0) mcnt += m;
#pragma unroll
        for (int j = 0; j < 8; ++j) {
            int c = lane + 32 * j;
            pacc[j] += ((v[j] - mu) * inv * nw[c] + nb[c]) * m;
        }
    }
#pragma unroll
    for (int j = 0; j < 8; ++j) pool_w[wave][lane + 32 * j] = pacc[j];
    if (lane == 0) cnt_w[wave] = mcnt;
    __syncthreads();

    int c = threadIdx.x;
    float p = 0.f, cnt = 0.f;
#pragma unroll
    for (int w = 0; w < 8; ++w) { p += pool_w[w][c]; cnt += cnt_w[w]; }
    p /= fmaxf(cnt, 1.f);

    for (int j = 0; j < 3; ++j) {
        red[c] = p * bw[j * DM + c];
        __syncthreads();
        for (int off = 128; off; off >>= 1) {
            if (c < off) red[c] += red[c + off];
            __syncthreads();
        }
        if (c == 0) bind_s[j] = red[0];
        __syncthreads();
    }
    if (c < 3) {
        float x = bind_s[c] + bb[c];
        out[(size_t)b * 3 + c] = 1.f / (1.f + __expf(-x));
    }
}

// ---------------------------------------------------------------------------
// Weight prep
// ---------------------------------------------------------------------------
__global__ __launch_bounds__(256) void f32_to_f16_kernel(
    const float* __restrict__ in, _Float16* __restrict__ out, size_t n)
{
    size_t i = (size_t)blockIdx.x * 256 + threadIdx.x;
    if (i < n) out[i] = (_Float16)in[i];
}

__global__ __launch_bounds__(256) void dtw_pad_kernel(
    const float* __restrict__ w, _Float16* __restrict__ out)
{
    int tid = blockIdx.x * 256 + threadIdx.x;
    if (tid >= NLAY * DI * 32) return;
    int j = tid & 31, rn = tid >> 5;
    out[tid] = (j < DR_N) ? (_Float16)w[(size_t)rn * DR_N + j] : (_Float16)0.f;
}

// ---------------------------------------------------------------------------
static inline int cdiv(int a, int b) { return (a + b - 1) / b; }

static inline void launch_gemm(hipStream_t st, const _Float16* A, int lda,
                               const _Float16* W, int ldw, int M, int N, int K,
                               float* Cf, _Float16* Ch, int ldc,
                               const float* bias, int mode)
{
    int blocks = (M / BM) * cdiv(N, BN);
    wmma_gemm_kernel<<<blocks, 256, 0, st>>>(A, lda, W, ldw, M, N, K,
                                             Cf, Ch, ldc, bias, mode);
}

extern "C" void kernel_launch(void* const* d_in, const int* in_sizes, int n_in,
                              void* d_out, int out_size, void* d_ws, size_t ws_size,
                              hipStream_t stream)
{
    const int*   tok       = (const int*)d_in[0];
    const int*   mask      = (const int*)d_in[1];
    const float* emb       = (const float*)d_in[2];
    const float* conv_w    = (const float*)d_in[3];
    const float* bn_gamma  = (const float*)d_in[4];
    const float* bn_beta   = (const float*)d_in[5];
    const float* in_proj_w = (const float*)d_in[6];
    const float* conv1d_w  = (const float*)d_in[7];
    const float* conv1d_b  = (const float*)d_in[8];
    const float* x_proj_w  = (const float*)d_in[9];
    const float* dt_proj_w = (const float*)d_in[10];
    const float* dt_proj_b = (const float*)d_in[11];
    const float* A_log     = (const float*)d_in[12];
    const float* Dp        = (const float*)d_in[13];
    const float* out_proj_w= (const float*)d_in[14];
    const float* norm1_w   = (const float*)d_in[15];
    const float* norm2_w   = (const float*)d_in[16];
    const float* fc1_w     = (const float*)d_in[17];
    const float* fc2_w     = (const float*)d_in[18];
    const float* normf_w   = (const float*)d_in[19];
    const float* normf_b   = (const float*)d_in[20];
    const float* bind_w    = (const float*)d_in[21];
    const float* bind_b    = (const float*)d_in[22];
    float* out = (float*)d_out;

    char* ws = (char*)d_ws;
    size_t off = 0;
    auto alloc = [&](size_t bytes) -> void* {
        size_t a = (off + 255) & ~(size_t)255;
        off = a + bytes;
        return (void*)(ws + a);
    };

    // f16 weights (all layers, converted once per launch)
    _Float16* w_in  = (_Float16*)alloc((size_t)NLAY * 1024 * 256 * 2);
    _Float16* w_xp  = (_Float16*)alloc((size_t)NLAY * 48   * 512 * 2);
    _Float16* w_dt  = (_Float16*)alloc((size_t)NLAY * 512  * 32  * 2);
    _Float16* w_out = (_Float16*)alloc((size_t)NLAY * 256  * 512 * 2);
    _Float16* w_fc1 = (_Float16*)alloc((size_t)NLAY * 256  * 256 * 2);
    _Float16* w_fc2 = (_Float16*)alloc((size_t)NLAY * 256  * 128 * 2);

    // per-chunk activations
    float*    hidden  = (float*)alloc((size_t)RCH * DM * 4);
    float*    resid   = (float*)alloc((size_t)RCH * DM * 4);
    _Float16* hs16    = (_Float16*)alloc((size_t)RCH * DM * 2);
    _Float16* xz16    = (_Float16*)alloc((size_t)RCH * 2 * DI * 2);
    float*    xc      = (float*)alloc((size_t)RCH * DI * 4);
    _Float16* xc16    = (_Float16*)alloc((size_t)RCH * DI * 2);
    float*    xdbl    = (float*)alloc((size_t)RCH * 48 * 4);
    _Float16* dtin16  = (_Float16*)alloc((size_t)RCH * 32 * 2);
    float*    dtbuf   = (float*)alloc((size_t)RCH * DI * 4);
    _Float16* y16     = (_Float16*)alloc((size_t)RCH * DI * 2);
    float*    yg      = (float*)alloc((size_t)RCH * 256 * 4);
    _Float16* h16     = (_Float16*)alloc((size_t)RCH * MLPH * 2);

    auto cvt = [&](const float* src, _Float16* dst, size_t n) {
        f32_to_f16_kernel<<<(unsigned)((n + 255) / 256), 256, 0, stream>>>(src, dst, n);
    };
    cvt(in_proj_w,  w_in,  (size_t)NLAY * 1024 * 256);
    cvt(x_proj_w,   w_xp,  (size_t)NLAY * 48   * 512);
    cvt(out_proj_w, w_out, (size_t)NLAY * 256  * 512);
    cvt(fc1_w,      w_fc1, (size_t)NLAY * 256  * 256);
    cvt(fc2_w,      w_fc2, (size_t)NLAY * 256  * 128);
    dtw_pad_kernel<<<cdiv(NLAY * DI * 32, 256), 256, 0, stream>>>(dt_proj_w, w_dt);

    for (int chunk = 0; chunk < NCHUNK; ++chunk) {
        int row_base = chunk * RCH;
        embed_conv_kernel<<<cdiv(RCH * DM, 256), 256, 0, stream>>>(
            tok, emb, conv_w, bn_gamma, bn_beta, hidden, resid, row_base);

        for (int i = 0; i < NLAY; ++i) {
            if (i > 0)
                add_kernel<<<cdiv(RCH * DM, 256), 256, 0, stream>>>(resid, hidden, RCH * DM);

            rmsnorm_f16_kernel<<<RCH / 8, 256, 0, stream>>>(resid, norm1_w + i * DM, hs16, RCH);

            // in_proj: (R x 256) x (1024 x 256)^T -> xz f16
            launch_gemm(stream, hs16, DM, w_in + (size_t)i * 1024 * 256, DM,
                        RCH, 2 * DI, DM, nullptr, xz16, 2 * DI, nullptr, 1);

            conv1d_silu_kernel<<<cdiv(RCH * DI, 256), 256, 0, stream>>>(
                xz16, conv1d_w, conv1d_b, xc, xc16, i);

            // x_proj: (R x 512) x (48 x 512)^T -> xdbl f32
            launch_gemm(stream, xc16, DI, w_xp + (size_t)i * 48 * 512, DI,
                        RCH, 48, DI, xdbl, nullptr, 48, nullptr, 0);

            dt_prep_kernel<<<cdiv(RCH * 32, 256), 256, 0, stream>>>(xdbl, dtin16, RCH);

            // dt_proj (+bias, softplus): (R x 32pad) x (512 x 32pad)^T -> dt f32
            launch_gemm(stream, dtin16, 32, w_dt + (size_t)i * 512 * 32, 32,
                        RCH, DI, 32, dtbuf, nullptr, DI, dt_proj_b + i * DI, 3);

            scan_kernel<<<cdiv(CB * DI, 256), 256, 0, stream>>>(
                dtbuf, xc, xdbl, xz16, A_log, Dp, y16, i);

            // out_proj: residual += (R x 512) x (256 x 512)^T
            launch_gemm(stream, y16, DI, w_out + (size_t)i * 256 * 512, DI,
                        RCH, DM, DI, resid, nullptr, DM, nullptr, 2);

            rmsnorm_f16_kernel<<<RCH / 8, 256, 0, stream>>>(resid, norm2_w + i * DM, hs16, RCH);

            // fc1: (R x 256) x (256 x 256)^T -> yg f32
            launch_gemm(stream, hs16, DM, w_fc1 + (size_t)i * 256 * 256, DM,
                        RCH, 256, DM, yg, nullptr, 256, nullptr, 0);

            swiglu_kernel<<<cdiv(RCH * MLPH, 256), 256, 0, stream>>>(yg, h16, RCH);

            // fc2: (R x 128) x (256 x 128)^T -> hidden f32
            launch_gemm(stream, h16, MLPH, w_fc2 + (size_t)i * 256 * 128, MLPH,
                        RCH, DM, MLPH, hidden, nullptr, DM, nullptr, 0);
        }

        final_pool_kernel<<<CB, 256, 0, stream>>>(
            hidden, resid, mask, normf_w, normf_b, bind_w, bind_b, out, chunk);
    }
}
